// YOLO_78907139162563
// MI455X (gfx1250) — matrix operand order repaired and checked by reading, not compiled, not run on hardware
//
#include <hip/hip_runtime.h>
#include <hip/hip_bf16.h>

// ---------------------------------------------------------------------------
// MI455X (gfx1250) YOLO forward: implicit-GEMM convs on v_wmma_f32_16x16x32_f16
// 128x64x32 block tile, 8 waves, 4 WMMAs/wave/K-step, division-free im2col,
// double-buffered LDS with software-pipelined async global->LDS staging
// (ASYNCcnt): stage tile t+1 while WMMAs consume tile t.
// ---------------------------------------------------------------------------

typedef __attribute__((ext_vector_type(16))) _Float16 v16h;
typedef __attribute__((ext_vector_type(8)))  _Float16 v8h;
typedef __attribute__((ext_vector_type(8)))  float    v8f;

#define BM 128
#define BN 64
#define BK 32
#define LDS_PAD 8   // 16B of halves: keeps 16B alignment, staggers banks

#define ALPHA_LEAKY 0.3f
#define BN_EPS_F    1e-3f
#define NBOX        1200
#define MAXB        100

struct ConvArgs {
  const _Float16* __restrict__ in;   // fp16 activations, NHWC (channel-strided view)
  const _Float16* __restrict__ w;    // fp16 weights, [K, Cout] (= HWIO flattened)
  const float* __restrict__ bias;
  const float* __restrict__ gamma;
  const float* __restrict__ beta;
  const float* __restrict__ mean;
  const float* __restrict__ var;
  _Float16* __restrict__ out;        // fp16 out (may be null)
  float*    __restrict__ out32;      // fp32 out (may be null, head only)
  int B, H, W, Cin, inStride, inOff;
  int OH, OW, Cout, outStride, outOff;
  int KH, KW, stride, padH, padW;
  int M;
  int bn, act;
};

__device__ __forceinline__ v16h ldFragA(const _Float16* p) {
  // A 16x32 fragment: lane group g holds K = {g*8..g*8+7, g*8+16..g*8+23}
  v8h lo = *(const v8h*)p;
  v8h hi = *(const v8h*)(p + 16);
  return __builtin_shufflevector(lo, hi, 0,1,2,3,4,5,6,7,8,9,10,11,12,13,14,15);
}
__device__ __forceinline__ v16h ldFragB(const _Float16* p) {
  // B 32x16 fragment: lane group g holds K = {g*8..g*8+7, g*8+16..g*8+23} of its column
  v8h lo = *(const v8h*)p;
  v8h hi = *(const v8h*)(p + 16);
  return __builtin_shufflevector(lo, hi, 0,1,2,3,4,5,6,7,8,9,10,11,12,13,14,15);
}
__device__ __forceinline__ v8f wmma_step(v16h fa, v16h fb, v8f c) {
  return __builtin_amdgcn_wmma_f32_16x16x32_f16(
      /*neg_a=*/false, fa, /*neg_b=*/false, fb,
      /*c_mod=*/(short)0, c, /*reuse_a=*/false, /*reuse_b=*/false);
}

// per-lane async global->LDS 16B copy (gfx1250, tracked by ASYNCcnt).
// VDST operand carries the LDS byte address (aperture maps addr[31:0] -> LDS).
__device__ __forceinline__ void async_copy_b128(const _Float16* gsrc, _Float16* lds_dst) {
  unsigned lds = (unsigned)(uintptr_t)lds_dst;
  unsigned long long ga = (unsigned long long)(uintptr_t)gsrc;
  asm volatile("global_load_async_to_lds_b128 %0, %1, off"
               :: "v"(lds), "v"(ga) : "memory");
}

__global__ __launch_bounds__(256) void conv_wmma(ConvArgs a) {
  __shared__ _Float16 As[2][BM][BK + LDS_PAD];    // [buf][m][k]
  __shared__ _Float16 Bst[2][BN][BK + LDS_PAD];   // [buf][n][k] (transposed weights)

  const int tid  = threadIdx.x;
  const int lane = tid & 31;
  const int wid  = tid >> 5;       // 8 waves
  const int wm   = wid & 3;        // M slab base wm*32
  const int wn   = wid >> 2;       // N slab base wn*32
  const int l16  = lane & 15;
  const int grp  = lane >> 4;      // half-wave group (ISA fragment layout)

  const int tileM = blockIdx.x * BM;
  const int tileN = blockIdx.y * BN;

  // A-staging map: row = tid/2 (0..127), colBase = (tid&1)*16
  const int arow = tid >> 1;
  const int acol = (tid & 1) * 16;
  const int m = tileM + arow;
  const bool mvalid = (m < a.M);
  int bI = 0, oh = 0, ow = 0;
  {
    int mm = mvalid ? m : 0;
    int hw = a.OH * a.OW;
    bI = mm / hw;
    int r = mm - bI * hw;
    oh = r / a.OW;
    ow = r - oh * a.OW;
  }
  const int ihBase = oh * a.stride - a.padH;
  const int iwBase = ow * a.stride - a.padW;

  // B-staging map: kr = tid/8 (0..31), nb = (tid&7)*8
  const int kr = tid >> 3;
  const int nb = (tid & 7) * 8;

  // Layer-uniform fast paths (scalar branches)
  const bool vecA = ((a.Cin & 31) == 0) && ((a.inStride & 15) == 0) && ((a.inOff & 15) == 0);
  const bool vecB = ((a.Cout & 7) == 0);

  // Stage one (kh,kw,c0) K-slice into LDS buffer `buf`.
  auto stageTile = [&](int skh, int skw, int sc0, int buf) {
    const int ih = ihBase + skh;
    const int iw = iwBase + skw;
    const bool vSp = mvalid && (ih >= 0) && (ih < a.H) && (iw >= 0) && (iw < a.W);
    const _Float16* __restrict__ pA =
        a.in + ((size_t)(bI * a.H + ih) * a.W + iw) * a.inStride + a.inOff;
    // ---- A ----
    if (vecA) {
      if (vSp) {
        async_copy_b128(pA + sc0 + acol,     &As[buf][arow][acol]);
        async_copy_b128(pA + sc0 + acol + 8, &As[buf][arow][acol + 8]);
      } else {
        v8h z = {};
        *(v8h*)&As[buf][arow][acol]     = z;
        *(v8h*)&As[buf][arow][acol + 8] = z;
      }
    } else {
#pragma unroll
      for (int j = 0; j < 16; ++j) {
        int ci = sc0 + acol + j;
        _Float16 v = (_Float16)0.0f;
        if (vSp && ci < a.Cin) v = pA[ci];
        As[buf][arow][acol + j] = v;
      }
    }
    // ---- B (transposed: Bst[n][k]) ----
    const int kbase = (skh * a.KW + skw) * a.Cin;
    const int kc = sc0 + kr;
    const size_t wrow = (size_t)(kbase + kc) * a.Cout + tileN + nb;
    if (vecB) {
      v8h d = {};
      if (kc < a.Cin && tileN + nb < a.Cout) d = *(const v8h*)(a.w + wrow);
#pragma unroll
      for (int j = 0; j < 8; ++j) Bst[buf][nb + j][kr] = d[j];
    } else {
#pragma unroll
      for (int j = 0; j < 8; ++j) {
        _Float16 v = (_Float16)0.0f;
        if (kc < a.Cin && (tileN + nb + j) < a.Cout) v = a.w[wrow + j];
        Bst[buf][nb + j][kr] = v;
      }
    }
    // prefetch the following weight K-slice into L2 (global_prefetch_b8)
    if (sc0 + BK < a.Cin)
      __builtin_prefetch(a.w + (size_t)(kbase + sc0 + BK) * a.Cout + tileN, 0, 0);
  };

  v8f acc00 = {}, acc01 = {}, acc10 = {}, acc11 = {};

  const int nC = (a.Cin + BK - 1) / BK;
  const int T  = a.KH * a.KW * nC;

  // next-stage indices, advanced with carry logic (no div/rem in the loop)
  int skh = 0, skw = 0, sc0 = 0;
  auto advance = [&]() {
    sc0 += BK;
    if (sc0 >= a.Cin) {
      sc0 = 0;
      if (++skw == a.KW) { skw = 0; ++skh; }
    }
  };

  // prologue: stage tile 0 into buffer 0
  stageTile(skh, skw, sc0, 0);
  advance();

  for (int t = 0; t < T; ++t) {
    const int cur = t & 1;
    // all waves finished reading buffer cur^1 (iteration t-1) -> safe to refill
    __syncthreads();
    if (t + 1 < T) {
      stageTile(skh, skw, sc0, cur ^ 1);
      advance();
      // loads retire in order: <=2 outstanding means tile t's copies landed
      if (vecA) asm volatile("s_wait_asynccnt 0x2" ::: "memory");
    } else {
      if (vecA) asm volatile("s_wait_asynccnt 0x0" ::: "memory");
    }
    __syncthreads();   // publish tile t to all waves

    // ---- 2x2 register block: 4 WMMAs reusing 2 A + 2 B fragments ---------
    v16h fa0 = ldFragA(&As[cur][wm * 32 + l16][grp * 8]);
    v16h fa1 = ldFragA(&As[cur][wm * 32 + 16 + l16][grp * 8]);
    v16h fb0 = ldFragB(&Bst[cur][wn * 32 + l16][grp * 8]);
    v16h fb1 = ldFragB(&Bst[cur][wn * 32 + 16 + l16][grp * 8]);
    acc00 = wmma_step(fa0, fb0, acc00);
    acc01 = wmma_step(fa0, fb1, acc01);
    acc10 = wmma_step(fa1, fb0, acc10);
    acc11 = wmma_step(fa1, fb1, acc11);
  }

  // ---- fused epilogue: bias + BN + leaky, fp16 (and/or fp32) store -------
#pragma unroll
  for (int im = 0; im < 2; ++im) {
#pragma unroll
    for (int in2 = 0; in2 < 2; ++in2) {
      v8f accv = (im == 0) ? (in2 == 0 ? acc00 : acc01)
                           : (in2 == 0 ? acc10 : acc11);
      const int n = tileN + wn * 32 + in2 * 16 + l16;
      if (n < a.Cout) {
        float bs = a.bias ? a.bias[n] : 0.0f;
        float sc = 1.0f, sh = 0.0f;
        if (a.bn) {
          float inv = rsqrtf(a.var[n] + BN_EPS_F) * a.gamma[n];
          sc = inv;
          sh = a.beta[n] - a.mean[n] * inv;
        }
#pragma unroll
        for (int r2 = 0; r2 < 8; ++r2) {
          int mg = tileM + wm * 32 + im * 16 + grp * 8 + r2;  // C/D: VGPR r, grp g -> M=g*8+r
          if (mg < a.M) {
            float v = accv[r2] + bs;
            if (a.bn) v = v * sc + sh;
            if (a.act) v = (v >= 0.0f) ? v : ALPHA_LEAKY * v;
            size_t o = (size_t)mg * a.outStride + a.outOff + n;
            if (a.out)   a.out[o]   = (_Float16)v;
            if (a.out32) a.out32[o] = v;
          }
        }
      }
    }
  }
}

// ---------------------------------------------------------------------------
__global__ void f2h_kernel(const float* __restrict__ in, _Float16* __restrict__ out, int n) {
  int t = blockIdx.x * blockDim.x + threadIdx.x;
  if (t < n) out[t] = (_Float16)in[t];
}

__global__ void copych_kernel(const _Float16* __restrict__ in, _Float16* __restrict__ out,
                              int M, int C, int is, int io, int os, int oo) {
  int t = blockIdx.x * blockDim.x + threadIdx.x;
  if (t >= M * C) return;
  int c = t % C, mm = t / C;
  out[(size_t)mm * os + oo + c] = in[(size_t)mm * is + io + c];
}

// 2x2 max pool, stride 1, SAME (pad_hi only) on a channel-sliced NHWC tensor
__global__ void pool2_kernel(const _Float16* __restrict__ in, _Float16* __restrict__ out,
                             int B, int H, int W, int C, int is, int io, int os, int oo) {
  int t = blockIdx.x * blockDim.x + threadIdx.x;
  if (t >= B * H * W * C) return;
  int c = t % C, r = t / C;
  int x = r % W; r /= W;
  int y = r % H; int b = r / H;
  auto ld = [&](int yy, int xx) {
    return (float)in[((size_t)(b * H + yy) * W + xx) * is + io + c];
  };
  float v = ld(y, x);
  if (x + 1 < W) v = fmaxf(v, ld(y, x + 1));
  if (y + 1 < H) {
    v = fmaxf(v, ld(y + 1, x));
    if (x + 1 < W) v = fmaxf(v, ld(y + 1, x + 1));
  }
  out[((size_t)(b * H + y) * W + x) * os + oo + c] = (_Float16)v;
}

// head decode: fp32 [B,20,20,21] -> boxes [B,1200,4], scores [B,1200]
__global__ void decode_kernel(const float* __restrict__ head,
                              float* __restrict__ boxes, float* __restrict__ scores) {
  int t = blockIdx.x * blockDim.x + threadIdx.x;
  const int G = 20, A = 3;
  if (t >= 4 * G * G * A) return;
  int a = t % A; int r = t / A;
  int j = r % G; r /= G;
  int i = r % G; int b = r / G;
  const float* p = &head[(((size_t)(b * G + i) * G) + j) * 21 + a * 7];
  float sx = 1.0f / (1.0f + expf(-p[0]));
  float sy = 1.0f / (1.0f + expf(-p[1]));
  float cx = (sx + (float)j) / (float)G;
  float cy = (sy + (float)i) / (float)G;
  const float AW[3] = {0.08f, 0.15f, 0.30f};
  const float AH[3] = {0.10f, 0.20f, 0.35f};
  float w = expf(p[2]) * AW[a];
  float h = expf(p[3]) * AH[a];
  float obj = 1.0f / (1.0f + expf(-p[4]));
  float c0 = 1.0f / (1.0f + expf(-p[5]));
  float c1 = 1.0f / (1.0f + expf(-p[6]));
  float cls = (c1 > c0) ? 1.0f : 0.0f;   // argmax index as float (ties -> 0)
  int n = (i * G + j) * A + a;
  float* bx = &boxes[((size_t)b * NBOX + n) * 4];
  bx[0] = cx - 0.5f * w; bx[1] = cy - 0.5f * h;
  bx[2] = cx + 0.5f * w; bx[3] = cy + 0.5f * h;
  scores[b * NBOX + n] = obj * cls;
}

// one workgroup per batch image: stable sort by score, O(N^2) suppression scan
__global__ __launch_bounds__(256) void nms_kernel(const float* __restrict__ boxes,
                                                  const float* __restrict__ scores,
                                                  float* __restrict__ outPred,
                                                  float* __restrict__ outValid) {
  __shared__ float ss0[NBOX];
  __shared__ float ssc[NBOX];
  __shared__ float sbx[NBOX][4];
  __shared__ int   sup[NBOX];
  const int b = blockIdx.x, tid = threadIdx.x;
  const float* sc = &scores[(size_t)b * NBOX];
  const float* bx = &boxes[(size_t)b * NBOX * 4];

  for (int i = tid; i < NBOX; i += 256) ss0[i] = sc[i];
  __syncthreads();

  // stable descending rank (matches jnp.argsort(-scores))
  for (int i = tid; i < NBOX; i += 256) {
    float si = ss0[i];
    int rank = 0;
    for (int j = 0; j < NBOX; ++j) {
      float sj = ss0[j];
      rank += (sj > si) || (sj == si && j < i);
    }
    ssc[rank] = si;
    sbx[rank][0] = bx[i * 4 + 0];
    sbx[rank][1] = bx[i * 4 + 1];
    sbx[rank][2] = bx[i * 4 + 2];
    sbx[rank][3] = bx[i * 4 + 3];
  }
  __syncthreads();
  for (int i = tid; i < NBOX; i += 256) sup[i] = !(ssc[i] > 0.25f);
  __syncthreads();

  for (int i = 0; i < NBOX - 1; ++i) {
    if (!sup[i]) {
      float x1 = sbx[i][0], y1 = sbx[i][1], x2 = sbx[i][2], y2 = sbx[i][3];
      float ai = (x2 - x1) * (y2 - y1);
      for (int j = i + 1 + tid; j < NBOX; j += 256) {
        float u1 = sbx[j][0], v1 = sbx[j][1], u2 = sbx[j][2], v2 = sbx[j][3];
        float aj = (u2 - u1) * (v2 - v1);
        float iw = fminf(x2, u2) - fmaxf(x1, u1); iw = iw > 0.0f ? iw : 0.0f;
        float ih = fminf(y2, v2) - fmaxf(y1, v1); ih = ih > 0.0f ? ih : 0.0f;
        float inter = iw * ih;
        float iou = inter / (ai + aj - inter + 1e-9f);
        if (iou > 0.5f) sup[j] = 1;
      }
    }
    __syncthreads();
  }

  if (tid == 0) {
    float* pr = &outPred[(size_t)b * MAXB * 6];
    int cnt = 0;
    for (int i = 0; i < NBOX && cnt < MAXB; ++i) {
      if (!sup[i]) {
        float* row = &pr[cnt * 6];
        row[0] = fminf(fmaxf(sbx[i][0], 0.0f), 1.0f);
        row[1] = fminf(fmaxf(sbx[i][1], 0.0f), 1.0f);
        row[2] = fminf(fmaxf(sbx[i][2], 0.0f), 1.0f);
        row[3] = fminf(fmaxf(sbx[i][3], 0.0f), 1.0f);
        row[4] = ssc[i];
        row[5] = 0.0f;
        ++cnt;
      }
    }
    for (int r = cnt; r < MAXB; ++r) {
      float* row = &pr[r * 6];
      row[0] = row[1] = row[2] = row[3] = row[4] = row[5] = 0.0f;
    }
    outValid[b] = (float)cnt;
  }
}

// ---------------------------------------------------------------------------
struct PSet { const float *W, *b, *g, *be, *mn, *vr; };

extern "C" void kernel_launch(void* const* d_in, const int* in_sizes, int n_in,
                              void* d_out, int out_size, void* d_ws, size_t ws_size,
                              hipStream_t stream) {
  (void)in_sizes; (void)n_in; (void)out_size; (void)ws_size;
  const int B = 4;

  // ---- parameter cursor over setup_inputs() dict order ----
  int cur = 1;  // d_in[0] = inputs
  auto take = [&](bool bn) {
    PSet p{};
    p.W = (const float*)d_in[cur++];
    p.b = (const float*)d_in[cur++];
    if (bn) {
      p.g  = (const float*)d_in[cur++];
      p.be = (const float*)d_in[cur++];
      p.mn = (const float*)d_in[cur++];
      p.vr = (const float*)d_in[cur++];
    }
    return p;
  };
  PSet p_conv1 = take(true), p_conv2 = take(true);
  PSet p_c2f1c = take(true), p_c2f1b1 = take(true), p_c2f1b2 = take(true);
  PSet p_conv3 = take(true);
  PSet p_c2f2c = take(true), p_c2f2b1 = take(true), p_c2f2b2 = take(true);
  PSet p_conv4 = take(true);
  PSet p_c2f3c = take(true), p_c2f3b1 = take(true), p_c2f3b2 = take(true);
  PSet p_conv5 = take(true);
  PSet p_c2f4c = take(true), p_c2f4b1 = take(true), p_c2f4b2 = take(true);
  PSet p_sppf  = take(false);
  PSet p_c2f8c = take(true), p_c2f8b1 = take(true), p_c2f8b2 = take(true);
  PSet p_headc = take(true);
  PSet p_heado = take(false);

  // ---- deterministic bump allocator over d_ws ----
  size_t wsoff = 0;
  auto walloc = [&](size_t bytes) -> void* {
    void* p = (char*)d_ws + wsoff;
    wsoff = (wsoff + bytes + 255) & ~(size_t)255;
    return p;
  };
  auto cvtW = [&](const float* W, int n) -> const _Float16* {
    _Float16* o = (_Float16*)walloc((size_t)n * 2);
    f2h_kernel<<<(n + 255) / 256, 256, 0, stream>>>(W, o, n);
    return o;
  };

  // ---- convert weights to fp16 once (deterministic) ----
  const _Float16* w_conv1 = cvtW(p_conv1.W, 3 * 3 * 3 * 64);
  const _Float16* w_conv2 = cvtW(p_conv2.W, 3 * 3 * 64 * 128);
  const _Float16* w_c2f1c = cvtW(p_c2f1c.W, 128 * 128);
  const _Float16* w_c2f1b1 = cvtW(p_c2f1b1.W, 3 * 3 * 64 * 32);
  const _Float16* w_c2f1b2 = cvtW(p_c2f1b2.W, 3 * 3 * 32 * 64);
  const _Float16* w_conv3 = cvtW(p_conv3.W, 3 * 3 * 128 * 256);
  const _Float16* w_c2f2c = cvtW(p_c2f2c.W, 256 * 256);
  const _Float16* w_c2f2b1 = cvtW(p_c2f2b1.W, 3 * 3 * 128 * 64);
  const _Float16* w_c2f2b2 = cvtW(p_c2f2b2.W, 3 * 3 * 64 * 128);
  const _Float16* w_conv4 = cvtW(p_conv4.W, 3 * 3 * 256 * 512);
  const _Float16* w_c2f3c = cvtW(p_c2f3c.W, 512 * 512);
  const _Float16* w_c2f3b1 = cvtW(p_c2f3b1.W, 3 * 3 * 256 * 128);
  const _Float16* w_c2f3b2 = cvtW(p_c2f3b2.W, 3 * 3 * 128 * 256);
  const _Float16* w_conv5 = cvtW(p_conv5.W, 3 * 3 * 512 * 512);
  const _Float16* w_c2f4c = cvtW(p_c2f4c.W, 512 * 512);
  const _Float16* w_c2f4b1 = cvtW(p_c2f4b1.W, 3 * 3 * 256 * 128);
  const _Float16* w_c2f4b2 = cvtW(p_c2f4b2.W, 3 * 3 * 128 * 256);
  const _Float16* w_sppf  = cvtW(p_sppf.W, 512 * 512);
  const _Float16* w_c2f8c = cvtW(p_c2f8c.W, 512 * 512);
  const _Float16* w_c2f8b1 = cvtW(p_c2f8b1.W, 3 * 3 * 256 * 128);
  const _Float16* w_c2f8b2 = cvtW(p_c2f8b2.W, 3 * 3 * 128 * 256);
  const _Float16* w_headc = cvtW(p_headc.W, 3 * 3 * 512 * 512);
  const _Float16* w_heado = cvtW(p_heado.W, 512 * 21);

  // ---- conv launch helper ----
  auto conv = [&](const _Float16* in, int H, int W, int Cin, int inStride, int inOff,
                  const _Float16* w16, const PSet& p, bool bn, bool act,
                  int KH, int KW, int s, int Cout,
                  _Float16* out, int outStride, int outOff, float* out32) {
    ConvArgs a{};
    a.in = in; a.w = w16;
    a.bias = p.b; a.gamma = p.g; a.beta = p.be; a.mean = p.mn; a.var = p.vr;
    a.out = out; a.out32 = out32;
    a.B = B; a.H = H; a.W = W; a.Cin = Cin; a.inStride = inStride; a.inOff = inOff;
    a.OH = (H + s - 1) / s; a.OW = (W + s - 1) / s;
    a.Cout = Cout; a.outStride = outStride; a.outOff = outOff;
    a.KH = KH; a.KW = KW; a.stride = s;
    int ptH = (a.OH - 1) * s + KH - H; a.padH = ptH > 0 ? ptH / 2 : 0;
    int ptW = (a.OW - 1) * s + KW - W; a.padW = ptW > 0 ? ptW / 2 : 0;
    a.M = B * a.OH * a.OW;
    a.bn = bn ? 1 : 0; a.act = act ? 1 : 0;
    dim3 g((a.M + BM - 1) / BM, (Cout + BN - 1) / BN);
    conv_wmma<<<g, 256, 0, stream>>>(a);
  };
  auto copych = [&](const _Float16* in, _Float16* out, int M, int C,
                    int is, int io, int os, int oo) {
    int total = M * C;
    copych_kernel<<<(total + 255) / 256, 256, 0, stream>>>(in, out, M, C, is, io, os, oo);
  };
  auto c2f = [&](const _Float16* in, int H, int W, int f,
                 const _Float16* wC, const _Float16* wB1, const _Float16* wB2,
                 const PSet& pc, const PSet& p1, const PSet& p2, _Float16* out) {
    size_t hw = (size_t)B * H * W;
    _Float16* y  = (_Float16*)walloc(hw * f * 2);
    _Float16* a1 = (_Float16*)walloc(hw * (f / 4) * 2);
    _Float16* y2 = (_Float16*)walloc(hw * f * 2);
    conv(in, H, W, f, f, 0, wC, pc, true, true, 1, 1, 1, f, y, f, 0, nullptr);
    conv(y, H, W, f / 2, f, 0, wB1, p1, true, true, 3, 3, 1, f / 4, a1, f / 4, 0, nullptr);
    copych(y, y2, (int)hw, f / 2, f, f / 2, f, 0);              // b -> y2[0 : f/2]
    conv(a1, H, W, f / 4, f / 4, 0, wB2, p2, true, true, 3, 3, 1, f / 2, y2, f, f / 2, nullptr);
    conv(y2, H, W, f, f, 0, wC, pc, true, true, 1, 1, 1, f, out, f, 0, nullptr);
  };

  // ---- forward pass ----
  const float* inputs = (const float*)d_in[0];
  size_t nIn = (size_t)B * 640 * 640 * 3;
  _Float16* in16 = (_Float16*)walloc(nIn * 2);
  f2h_kernel<<<(int)((nIn + 255) / 256), 256, 0, stream>>>(inputs, in16, (int)nIn);

  _Float16* x1 = (_Float16*)walloc((size_t)B * 320 * 320 * 64 * 2);
  conv(in16, 640, 640, 3, 3, 0, w_conv1, p_conv1, true, true, 3, 3, 2, 64, x1, 64, 0, nullptr);

  _Float16* x2 = (_Float16*)walloc((size_t)B * 160 * 160 * 128 * 2);
  conv(x1, 320, 320, 64, 64, 0, w_conv2, p_conv2, true, true, 3, 3, 2, 128, x2, 128, 0, nullptr);

  _Float16* x2b = (_Float16*)walloc((size_t)B * 160 * 160 * 128 * 2);
  c2f(x2, 160, 160, 128, w_c2f1c, w_c2f1b1, w_c2f1b2, p_c2f1c, p_c2f1b1, p_c2f1b2, x2b);

  _Float16* x3 = (_Float16*)walloc((size_t)B * 80 * 80 * 256 * 2);
  conv(x2b, 160, 160, 128, 128, 0, w_conv3, p_conv3, true, true, 3, 3, 2, 256, x3, 256, 0, nullptr);
  _Float16* x3b = (_Float16*)walloc((size_t)B * 80 * 80 * 256 * 2);
  c2f(x3, 80, 80, 256, w_c2f2c, w_c2f2b1, w_c2f2b2, p_c2f2c, p_c2f2b1, p_c2f2b2, x3b);

  _Float16* x4 = (_Float16*)walloc((size_t)B * 40 * 40 * 512 * 2);
  conv(x3b, 80, 80, 256, 256, 0, w_conv4, p_conv4, true, true, 3, 3, 2, 512, x4, 512, 0, nullptr);
  _Float16* x4b = (_Float16*)walloc((size_t)B * 40 * 40 * 512 * 2);
  c2f(x4, 40, 40, 512, w_c2f3c, w_c2f3b1, w_c2f3b2, p_c2f3c, p_c2f3b1, p_c2f3b2, x4b);

  _Float16* x5 = (_Float16*)walloc((size_t)B * 20 * 20 * 512 * 2);
  conv(x4b, 40, 40, 512, 512, 0, w_conv5, p_conv5, true, true, 3, 3, 2, 512, x5, 512, 0, nullptr);
  _Float16* x5b = (_Float16*)walloc((size_t)B * 20 * 20 * 512 * 2);
  c2f(x5, 20, 20, 512, w_c2f4c, w_c2f4b1, w_c2f4b2, p_c2f4c, p_c2f4b1, p_c2f4b2, x5b);

  // ---- SPPF (20x20) ----
  int M20 = B * 20 * 20;
  _Float16* sy = (_Float16*)walloc((size_t)M20 * 512 * 2);
  conv(x5b, 20, 20, 512, 512, 0, w_sppf, p_sppf, false, false, 1, 1, 1, 512, sy, 512, 0, nullptr);
  _Float16* P1 = (_Float16*)walloc((size_t)M20 * 256 * 2);
  _Float16* P2 = (_Float16*)walloc((size_t)M20 * 128 * 2);
  _Float16* P3 = (_Float16*)walloc((size_t)M20 * 64 * 2);
  {
    int t1 = M20 * 256, t2 = M20 * 128, t3 = M20 * 64;
    pool2_kernel<<<(t1 + 255) / 256, 256, 0, stream>>>(sy, P1, B, 20, 20, 256, 512, 0, 256, 0);
    pool2_kernel<<<(t2 + 255) / 256, 256, 0, stream>>>(P1, P2, B, 20, 20, 128, 256, 0, 128, 0);
    pool2_kernel<<<(t3 + 255) / 256, 256, 0, stream>>>(P2, P3, B, 20, 20, 64, 128, 0, 64, 0);
  }
  _Float16* cat = (_Float16*)walloc((size_t)M20 * 512 * 2);
  copych(P3, cat, M20, 64, 64, 0, 512, 0);          // pooled y
  copych(sy, cat, M20, 256, 512, 256, 512, 64);     // x1
  copych(P1, cat, M20, 128, 256, 128, 512, 320);    // x2
  copych(P2, cat, M20, 64, 128, 64, 512, 448);      // x3
  _Float16* s5 = (_Float16*)walloc((size_t)M20 * 512 * 2);
  conv(cat, 20, 20, 512, 512, 0, w_sppf, p_sppf, false, false, 1, 1, 1, 512, s5, 512, 0, nullptr);

  _Float16* s6 = (_Float16*)walloc((size_t)M20 * 512 * 2);
  c2f(s5, 20, 20, 512, w_c2f8c, w_c2f8b1, w_c2f8b2, p_c2f8c, p_c2f8b1, p_c2f8b2, s6);

  _Float16* h1 = (_Float16*)walloc((size_t)M20 * 512 * 2);
  conv(s6, 20, 20, 512, 512, 0, w_headc, p_headc, true, true, 3, 3, 1, 512, h1, 512, 0, nullptr);

  float* headRaw = (float*)walloc((size_t)M20 * 21 * 4);
  conv(h1, 20, 20, 512, 512, 0, w_heado, p_heado, false, false, 1, 1, 1, 21,
       nullptr, 21, 0, headRaw);

  // ---- decode + NMS ----
  float* boxes  = (float*)walloc((size_t)B * NBOX * 4 * 4);
  float* scores = (float*)walloc((size_t)B * NBOX * 4);
  decode_kernel<<<(B * 20 * 20 * 3 + 255) / 256, 256, 0, stream>>>(headRaw, boxes, scores);

  float* outF = (float*)d_out;
  nms_kernel<<<B, 256, 0, stream>>>(boxes, scores, outF, outF + (size_t)B * MAXB * 6);
}
